// Decoder_83837761618616
// MI455X (gfx1250) — matrix operand order repaired and checked by reading, not compiled, not run on hardware
//
#include <hip/hip_runtime.h>
#include <math.h>

// ---------------------------------------------------------------------------
// Decoder (attention LSTM), B=128, L=512, KV=128, H=512, OUT=64
// 512 sequential stages; per stage two WMMA-bf16 gate GEMM kernels
// (fused LSTM cell epilogue) + one fused attention/softmax/context/pred kernel.
// Activations & weights stored bf16 (converted once per launch), f32 accum.
// ---------------------------------------------------------------------------

#define BATCH 128
#define SEQL  512
#define KVD   128
#define HID   512
#define OUTD  64

typedef __attribute__((ext_vector_type(16))) __bf16 v16bf;
typedef __attribute__((ext_vector_type(8)))  __bf16 v8bf;
typedef __attribute__((ext_vector_type(8)))  float  v8f;

static __device__ __forceinline__ v16bf ld_frag(const __bf16* p0, const __bf16* p1) {
  union { v16bf v; v8bf h[2]; } u;
  u.h[0] = *(const v8bf*)p0;
  u.h[1] = *(const v8bf*)p1;
  return u.v;
}

// Branchless fast sigmoid / tanh via v_exp_f32 (no libm EXEC divergence).
static __device__ __forceinline__ float sigm(float x) {
  return 1.0f / (1.0f + __expf(-x));
}
static __device__ __forceinline__ float tanh_fast(float x) {
  const float xc = fminf(fmaxf(x, -15.0f), 15.0f);   // saturate; exp stays finite
  const float e  = __expf(2.0f * xc);
  return (e - 1.0f) / (e + 1.0f);
}

// ---------------------------------------------------------------------------
// Fused gates GEMM (bf16 WMMA) + LSTM cell epilogue.
// gates(B,4*Hn) = X(B, Kin+Kh) @ [Wih|Whh]^T + bsum ; then i,f,g,o -> h,c.
// Each wave computes a 16(M) x 16(N) tile for ALL FOUR gates so the cell
// update is local. M=128 fixed -> 8 M-tiles; waves = 8 * (Hn/16).
// K loop split into ih / hh phases with pointer-bumped per-lane addresses.
// ---------------------------------------------------------------------------
__global__ __launch_bounds__(256) void lstm_gates_wmma(
    const __bf16* __restrict__ X, int ldx,
    const __bf16* __restrict__ Wih, int Kin,
    const __bf16* __restrict__ Whh, int Kh,
    const float*  __restrict__ bsum, int Hn,
    float*        __restrict__ cst,
    __bf16* __restrict__ outA, int ldA, int offA,
    __bf16* __restrict__ outB, int ldB, int offB,
    float*  __restrict__ outF, int ldF)
{
  const int wid   = (int)((blockIdx.x * blockDim.x + threadIdx.x) >> 5);
  const int lane  = (int)(threadIdx.x & 31u);
  const int mtile = wid & 7;        // 128/16 = 8 M tiles
  const int ntile = wid >> 3;       // Hn/16 N tiles (grid sized exactly)
  const int lh    = lane >> 4;      // lane half (0/1)
  const int lm    = lane & 15;
  const int m     = mtile * 16 + lm;      // A-matrix row owned by this lane
  const int nrow  = ntile * 16 + lm;      // hidden-unit (gate column) base

  v8f acc[4] = {};                  // f32 accumulators: i, f, g, o

  // A: per-lane base (lane-lo K {k..k+15}; lane-hi K {k+16..k+31})
  const __bf16* xp = X + (size_t)m * ldx + lh * 16;

  // ---- phase 1: input-to-hidden weights ----
  {
    const __bf16* w0 = Wih + (size_t)(0 * Hn + nrow) * Kin + lh * 16;
    const __bf16* w1 = Wih + (size_t)(1 * Hn + nrow) * Kin + lh * 16;
    const __bf16* w2 = Wih + (size_t)(2 * Hn + nrow) * Kin + lh * 16;
    const __bf16* w3 = Wih + (size_t)(3 * Hn + nrow) * Kin + lh * 16;
    for (int k = 0; k < Kin; k += 32) {
      const v16bf a = ld_frag(xp, xp + 8);
      acc[0] = __builtin_amdgcn_wmma_f32_16x16x32_bf16(false, a, false, ld_frag(w0, w0 + 8), (short)0, acc[0], false, false);
      acc[1] = __builtin_amdgcn_wmma_f32_16x16x32_bf16(false, a, false, ld_frag(w1, w1 + 8), (short)0, acc[1], false, false);
      acc[2] = __builtin_amdgcn_wmma_f32_16x16x32_bf16(false, a, false, ld_frag(w2, w2 + 8), (short)0, acc[2], false, false);
      acc[3] = __builtin_amdgcn_wmma_f32_16x16x32_bf16(false, a, false, ld_frag(w3, w3 + 8), (short)0, acc[3], false, false);
      xp += 32; w0 += 32; w1 += 32; w2 += 32; w3 += 32;
    }
  }
  // ---- phase 2: hidden-to-hidden weights ----
  {
    const __bf16* w0 = Whh + (size_t)(0 * Hn + nrow) * Kh + lh * 16;
    const __bf16* w1 = Whh + (size_t)(1 * Hn + nrow) * Kh + lh * 16;
    const __bf16* w2 = Whh + (size_t)(2 * Hn + nrow) * Kh + lh * 16;
    const __bf16* w3 = Whh + (size_t)(3 * Hn + nrow) * Kh + lh * 16;
    for (int k = 0; k < Kh; k += 32) {
      const v16bf a = ld_frag(xp, xp + 8);
      acc[0] = __builtin_amdgcn_wmma_f32_16x16x32_bf16(false, a, false, ld_frag(w0, w0 + 8), (short)0, acc[0], false, false);
      acc[1] = __builtin_amdgcn_wmma_f32_16x16x32_bf16(false, a, false, ld_frag(w1, w1 + 8), (short)0, acc[1], false, false);
      acc[2] = __builtin_amdgcn_wmma_f32_16x16x32_bf16(false, a, false, ld_frag(w2, w2 + 8), (short)0, acc[2], false, false);
      acc[3] = __builtin_amdgcn_wmma_f32_16x16x32_bf16(false, a, false, ld_frag(w3, w3 + 8), (short)0, acc[3], false, false);
      xp += 32; w0 += 32; w1 += 32; w2 += 32; w3 += 32;
    }
  }

  // Epilogue: D layout => VGPR r : lane<16 -> (M=r, N=lane) ; lane>=16 -> (M=r+8)
  const int nb = nrow;
  const float bi  = bsum[0 * Hn + nb];
  const float bf_ = bsum[1 * Hn + nb];
  const float bg  = bsum[2 * Hn + nb];
  const float bo  = bsum[3 * Hn + nb];

  #pragma unroll
  for (int r = 0; r < 8; ++r) {
    const int mm = mtile * 16 + lh * 8 + r;
    const float iv = sigm(acc[0][r] + bi);
    const float fv = sigm(acc[1][r] + bf_);
    const float gv = tanh_fast(acc[2][r] + bg);
    const float ov = sigm(acc[3][r] + bo);
    const size_t ci = (size_t)mm * Hn + nb;
    const float cn = fv * cst[ci] + iv * gv;
    cst[ci] = cn;
    const float hv = ov * tanh_fast(cn);
    if (outA) outA[(size_t)mm * ldA + offA + nb] = (__bf16)hv;
    if (outB) outB[(size_t)mm * ldB + offB + nb] = (__bf16)hv;
    if (outF) outF[(size_t)mm * ldF + nb] = hv;
  }
}

// ---------------------------------------------------------------------------
// Fused attention: energy = K_b @ q, masked softmax, ctx = score @ V_b,
// oc = [h2, ctx] -> bf16 into next x1 buffer; pred = oc @ Wout^T + bout.
// One block (256 threads) per batch row. q == nullptr means q = 0 (init step).
// ---------------------------------------------------------------------------
__global__ __launch_bounds__(256) void attn_kernel(
    const float* __restrict__ key, const float* __restrict__ value,
    const int*   __restrict__ enc, const float* __restrict__ q,
    __bf16*      __restrict__ ocdst,
    const float* __restrict__ W_out, const float* __restrict__ b_out,
    float*       __restrict__ pred)
{
  __shared__ float sc[SEQL];
  __shared__ float red[256];
  __shared__ float oc_s[2 * KVD];

  const int b = (int)blockIdx.x;
  const int tid = (int)threadIdx.x;

  if (tid < KVD) oc_s[tid] = q ? q[(size_t)b * KVD + tid] : 0.0f;
  __syncthreads();

  const int len = enc[b];

  // ---- energies (2 rows of l per thread) ----
  #pragma unroll
  for (int rep = 0; rep < 2; ++rep) {
    const int l = tid + rep * 256;
    float e = -INFINITY;
    if (l < len) {
      const float4* kr = (const float4*)(key + ((size_t)l * BATCH + b) * KVD);
      float s = 0.0f;
      #pragma unroll
      for (int j = 0; j < KVD / 4; ++j) {
        const float4 k4 = kr[j];
        s += k4.x * oc_s[4 * j + 0] + k4.y * oc_s[4 * j + 1] +
             k4.z * oc_s[4 * j + 2] + k4.w * oc_s[4 * j + 3];
      }
      e = s;
    }
    sc[l] = e;
  }
  __syncthreads();

  // ---- max reduce ----
  red[tid] = fmaxf(sc[tid], sc[tid + 256]);
  __syncthreads();
  for (int s2 = 128; s2 > 0; s2 >>= 1) {
    if (tid < s2) red[tid] = fmaxf(red[tid], red[tid + s2]);
    __syncthreads();
  }
  const float mx = red[0];
  __syncthreads();

  // ---- exp + sum reduce ----
  const float e0 = (sc[tid]       == -INFINITY) ? 0.0f : __expf(sc[tid]       - mx);
  const float e1 = (sc[tid + 256] == -INFINITY) ? 0.0f : __expf(sc[tid + 256] - mx);
  sc[tid] = e0; sc[tid + 256] = e1;
  red[tid] = e0 + e1;
  __syncthreads();
  for (int s2 = 128; s2 > 0; s2 >>= 1) {
    if (tid < s2) red[tid] += red[tid + s2];
    __syncthreads();
  }
  const float inv = 1.0f / red[0];
  __syncthreads();

  // ---- context: k = tid&127, l-half = tid>>7 ----
  const int k  = tid & (KVD - 1);
  const int l0 = (tid >> 7) * 256;
  float acc = 0.0f;
  for (int l = l0; l < l0 + 256; ++l) {
    __builtin_prefetch(value + ((size_t)(l + 8) * BATCH + b) * KVD + k, 0, 3);
    acc += sc[l] * value[((size_t)l * BATCH + b) * KVD + k];
  }
  red[tid] = acc;
  __syncthreads();
  if (tid < KVD) oc_s[KVD + tid] = (red[tid] + red[tid + 128]) * inv;
  __syncthreads();

  // ---- write bf16 oc into next-step x1 buffer (cols 0..255 of ld 768) ----
  ocdst[(size_t)b * (2 * KVD + HID) + tid] = (__bf16)oc_s[tid];

  // ---- output projection ----
  if (pred && tid < OUTD) {
    const float* wr = W_out + (size_t)tid * (2 * KVD);
    float s = b_out[tid];
    #pragma unroll 8
    for (int kk = 0; kk < 2 * KVD; ++kk) s += oc_s[kk] * wr[kk];
    pred[(size_t)b * OUTD + tid] = s;
  }
}

// ---------------------------------------------------------------------------
// Prep utilities
// ---------------------------------------------------------------------------
__global__ void cvt_f32_bf16(const float* __restrict__ s, __bf16* __restrict__ d, int n) {
  for (int i = (int)(blockIdx.x * blockDim.x + threadIdx.x); i < n;
       i += (int)(gridDim.x * blockDim.x))
    d[i] = (__bf16)s[i];
}
__global__ void vec_add(const float* __restrict__ a, const float* __restrict__ b,
                        float* __restrict__ o, int n) {
  for (int i = (int)(blockIdx.x * blockDim.x + threadIdx.x); i < n;
       i += (int)(gridDim.x * blockDim.x))
    o[i] = a[i] + b[i];
}
__global__ void zero_f32(float* __restrict__ p, int n) {
  for (int i = (int)(blockIdx.x * blockDim.x + threadIdx.x); i < n;
       i += (int)(gridDim.x * blockDim.x))
    p[i] = 0.0f;
}
__global__ void zero_bf16(__bf16* __restrict__ p, int n) {
  for (int i = (int)(blockIdx.x * blockDim.x + threadIdx.x); i < n;
       i += (int)(gridDim.x * blockDim.x))
    p[i] = (__bf16)0.0f;
}

// ---------------------------------------------------------------------------
extern "C" void kernel_launch(void* const* d_in, const int* in_sizes, int n_in,
                              void* d_out, int out_size, void* d_ws, size_t ws_size,
                              hipStream_t stream) {
  (void)in_sizes; (void)n_in; (void)out_size; (void)ws_size;

  const float* key   = (const float*)d_in[0];
  const float* value = (const float*)d_in[1];
  const int*   enc   = (const int*)d_in[2];
  const float* W_ih1 = (const float*)d_in[3];
  const float* W_hh1 = (const float*)d_in[4];
  const float* b_ih1 = (const float*)d_in[5];
  const float* b_hh1 = (const float*)d_in[6];
  const float* W_ih2 = (const float*)d_in[7];
  const float* W_hh2 = (const float*)d_in[8];
  const float* b_ih2 = (const float*)d_in[9];
  const float* b_hh2 = (const float*)d_in[10];
  const float* W_out = (const float*)d_in[11];
  const float* b_out = (const float*)d_in[12];
  float* preds = (float*)d_out;

  // ---- workspace carve-out (~5 MB) ----
  char* ws = (char*)d_ws;
  size_t off = 0;
  auto take = [&](size_t bytes) -> char* {
    char* p = ws + off;
    off = (off + bytes + 255) & ~(size_t)255;
    return p;
  };
  __bf16* wih1b = (__bf16*)take((size_t)4 * HID * (2 * KVD) * 2);   // 2048x256
  __bf16* whh1b = (__bf16*)take((size_t)4 * HID * HID * 2);         // 2048x512
  __bf16* wih2b = (__bf16*)take((size_t)4 * KVD * HID * 2);         // 512x512
  __bf16* whh2b = (__bf16*)take((size_t)4 * KVD * KVD * 2);         // 512x128
  float*  b1s   = (float*)take((size_t)4 * HID * 4);
  float*  b2s   = (float*)take((size_t)4 * KVD * 4);
  const int LD1 = 2 * KVD + HID;  // 768 : [oc(256) | h1(512)]
  const int LD2 = HID + KVD;      // 640 : [h1(512) | h2(128)]
  __bf16* x1[2] = {(__bf16*)take((size_t)BATCH * LD1 * 2),
                   (__bf16*)take((size_t)BATCH * LD1 * 2)};
  __bf16* x2[2] = {(__bf16*)take((size_t)BATCH * LD2 * 2),
                   (__bf16*)take((size_t)BATCH * LD2 * 2)};
  float* c1  = (float*)take((size_t)BATCH * HID * 4);
  float* c2  = (float*)take((size_t)BATCH * KVD * 4);
  float* h2f = (float*)take((size_t)BATCH * KVD * 4);

  // ---- one-time (per launch) prep: weight bf16 conversion, bias sums, zeros
  cvt_f32_bf16<<<256, 256, 0, stream>>>(W_ih1, wih1b, 4 * HID * (2 * KVD));
  cvt_f32_bf16<<<256, 256, 0, stream>>>(W_hh1, whh1b, 4 * HID * HID);
  cvt_f32_bf16<<<256, 256, 0, stream>>>(W_ih2, wih2b, 4 * KVD * HID);
  cvt_f32_bf16<<<128, 256, 0, stream>>>(W_hh2, whh2b, 4 * KVD * KVD);
  vec_add<<<8, 256, 0, stream>>>(b_ih1, b_hh1, b1s, 4 * HID);
  vec_add<<<2, 256, 0, stream>>>(b_ih2, b_hh2, b2s, 4 * KVD);
  zero_f32<<<64, 256, 0, stream>>>(c1, BATCH * HID);
  zero_f32<<<16, 256, 0, stream>>>(c2, BATCH * KVD);
  zero_bf16<<<96, 256, 0, stream>>>(x1[0], BATCH * LD1);
  zero_bf16<<<80, 256, 0, stream>>>(x2[0], BATCH * LD2);

  // ---- initial context: oc0 = [0, attend(0)] into x1[0] ----
  attn_kernel<<<BATCH, 256, 0, stream>>>(key, value, enc, nullptr, x1[0],
                                         W_out, b_out, nullptr);

  // ---- 512 sequential decoder steps ----
  for (int t = 0; t < SEQL; ++t) {
    const int cur = t & 1, nxt = cur ^ 1;
    // LSTM1: X = x1[cur] (B x 768) ; writes h1 bf16 into x2[cur][:, 0:512]
    // (consumed this step) and x1[nxt][:, 256:768] (consumed next step).
    lstm_gates_wmma<<<32, 256, 0, stream>>>(
        x1[cur], LD1, wih1b, 2 * KVD, whh1b, HID, b1s, HID, c1,
        x2[cur], LD2, 0, x1[nxt], LD1, 2 * KVD, nullptr, 0);
    // LSTM2: X = x2[cur] (B x 640) ; writes h2 bf16 into x2[nxt][:, 512:640]
    // and h2 f32 (attention query / oc head) into h2f.
    lstm_gates_wmma<<<8, 256, 0, stream>>>(
        x2[cur], LD2, wih2b, HID, whh2b, KVD, b2s, KVD, c2,
        x2[nxt], LD2, HID, nullptr, 0, 0, h2f, KVD);
    // Attention + oc pack (bf16 into x1[nxt][:, 0:256]) + pred.
    attn_kernel<<<BATCH, 256, 0, stream>>>(
        key, value, enc, h2f, x1[nxt], W_out, b_out,
        preds + (size_t)t * BATCH * OUTD);
  }
}
